// GatedGraphReasoning_89910845374721
// MI455X (gfx1250) — matrix operand order repaired and checked by reading, not compiled, not run on hardware
//
#include <hip/hip_runtime.h>
#include <cstddef>
#include <cstdint>

// ---------------------------------------------------------------------------
// Problem constants (fixed by the reference: B=2048, N=65536, D=512, k=5)
// ---------------------------------------------------------------------------
#define BB      2048
#define NN      65536
#define DD      512
#define KSEL    5
#define NCHUNK  32              // N split into 32 chunks of 2048 cols (128 tiles)
#define TILES_PER_CHUNK  (NN / (NCHUNK * 16))   // 128
#define KSTEPS  (DD / 32)       // 16 bf16 WMMA K-steps per 16x16 tile
#define TEMPF   10.0f

typedef __attribute__((ext_vector_type(16))) __bf16 v16bf;
typedef __attribute__((ext_vector_type(8)))  __bf16 v8bf;
typedef __attribute__((ext_vector_type(8)))  float  v8f;

// ---------------- bf16 helpers (bit-exact RNE, no native bf16 math) --------
__device__ __forceinline__ unsigned short f2bf_u(float f) {
  unsigned u = __float_as_uint(f);
  unsigned r = u + 0x7FFFu + ((u >> 16) & 1u);
  return (unsigned short)(r >> 16);
}
__device__ __forceinline__ float bf2f(unsigned short h) {
  return __uint_as_float(((unsigned)h) << 16);
}

// ---------------------------------------------------------------------------
// K0: per-row L2 normalize (divide by max(||r||, eps)) -> bf16, keep 1/||r||
// ---------------------------------------------------------------------------
__global__ void ggr_norm_rows(const float* __restrict__ src,
                              unsigned short* __restrict__ dstb,
                              float* __restrict__ invn, int rows) {
  int row  = blockIdx.x * 8 + (threadIdx.x >> 5);
  int lane = threadIdx.x & 31;
  if (row >= rows) return;
  const float* r = src + (size_t)row * DD + lane * 16;
  float v[16];
  float s = 0.f;
#pragma unroll
  for (int t = 0; t < 16; ++t) { v[t] = r[t]; s += v[t] * v[t]; }
#pragma unroll
  for (int o = 16; o >= 1; o >>= 1) s += __shfl_xor(s, o, 32);
  float inv = 1.0f / fmaxf(sqrtf(s), 1e-12f);
  if (lane == 0) invn[row] = inv;
  union { unsigned short u[16]; uint4 q[2]; } pk;
#pragma unroll
  for (int t = 0; t < 16; ++t) pk.u[t] = f2bf_u(v[t] * inv);
  uint4* d = (uint4*)(dstb + (size_t)row * DD + lane * 16);
  d[0] = pk.q[0];
  d[1] = pk.q[1];
}

// ---------------------------------------------------------------------------
// K1: split W (fp32) into bf16 hi + bf16 residual lo
// ---------------------------------------------------------------------------
__global__ void ggr_split_w(const float* __restrict__ W,
                            unsigned short* __restrict__ whi,
                            unsigned short* __restrict__ wlo) {
  int i = blockIdx.x * blockDim.x + threadIdx.x;
  float w = W[i];
  unsigned short h = f2bf_u(w);
  whi[i] = h;
  wlo[i] = f2bf_u(w - bf2f(h));
}

// ---------------- register top-5 insertion (val desc) ----------------------
__device__ __forceinline__ void top5_insert(float v, int id,
    float& v0, float& v1, float& v2, float& v3, float& v4,
    int& i0, int& i1, int& i2, int& i3, int& i4) {
  if (v <= v4) return;
  if (v > v0)      { v4=v3;i4=i3; v3=v2;i3=i2; v2=v1;i2=i1; v1=v0;i1=i0; v0=v;i0=id; }
  else if (v > v1) { v4=v3;i4=i3; v3=v2;i3=i2; v2=v1;i2=i1; v1=v;i1=id; }
  else if (v > v2) { v4=v3;i4=i3; v3=v2;i3=i2; v2=v;i2=id; }
  else if (v > v3) { v4=v3;i4=i3; v3=v;i3=id; }
  else             { v4=v;i4=id; }
}

// fragment loader: 16 contiguous K elements per lane (k = 16h + s)
__device__ __forceinline__ v16bf load_a_frag(const unsigned short* base) {
  union { v16bf v; v8bf h[2]; } u;
  u.h[0] = *(const v8bf*)(base);
  u.h[1] = *(const v8bf*)(base + 8);
  return u.v;
}

// ---------------------------------------------------------------------------
// K2: streamed sim GEMM + per-wave top-5.
//   grid = (NCHUNK, 16), block = 256 (8 waves).
//   All 8 waves of a block share the SAME N-tile stream (one chunk), each on
//   its own 16-row M-tile -> the 16KB B tile is loaded once per block into a
//   double-buffered LDS buffer (8x L2 traffic reduction) while the next tile
//   is prefetched through registers underneath the WMMA chain.
// ---------------------------------------------------------------------------
__global__ void __launch_bounds__(256)
ggr_sim_topk(const unsigned short* __restrict__ xnb,
             const unsigned short* __restrict__ mnb,
             float* __restrict__ cval, int* __restrict__ cidx) {
  __shared__ unsigned short bbuf[2][16 * DD];   // 2 x 16KB B tiles
  __shared__ float simtile[8][256];             // per-wave transpose scratch

  const int tid   = threadIdx.x;
  const int lane  = tid & 31;
  const int wid   = tid >> 5;
  const int chunk = blockIdx.x;
  const int mtile = blockIdx.y * 8 + wid;
  const int ncol0 = chunk * (TILES_PER_CHUNK * 16);

  // ---- preload A fragments for this wave's 16 rows, all 16 K-steps --------
  const int asub = lane & 16;                         // 0 or 16
  const unsigned short* arow =
      xnb + (size_t)(mtile * 16 + (lane & 15)) * DD + asub;
  v16bf afrag[KSTEPS];
#pragma unroll
  for (int kk = 0; kk < KSTEPS; ++kk) afrag[kk] = load_a_frag(arow + kk * 32);

  // ---- cooperative load of B tile 0 into bbuf[0] (16KB, fully contiguous) -
  {
    const uint4* src = (const uint4*)(mnb + (size_t)ncol0 * DD) + tid * 4;
    uint4* dst = (uint4*)&bbuf[0][0] + tid * 4;
    dst[0] = src[0]; dst[1] = src[1]; dst[2] = src[2]; dst[3] = src[3];
  }
  __syncthreads();

  float tv0=-3e38f, tv1=-3e38f, tv2=-3e38f, tv3=-3e38f, tv4=-3e38f;
  int   ti0=0, ti1=0, ti2=0, ti3=0, ti4=0;

  const int bsub   = lane & 16;                       // 0 or 16 (K half)
  float* tile      = &simtile[wid][0];
  const int n_lo   = lane & 15;
  const int m_hi   = (lane & 16) >> 1;                // 0 or 8 (C row half)
  const int m_own  = lane & 15;
  const int n_half = (lane >> 4) * 8;

  for (int t = 0; t < TILES_PER_CHUNK; ++t) {
    const int cur = t & 1;
    const int nxt = cur ^ 1;

    // ---- prefetch next B tile into registers (in flight under the WMMAs) --
    const int tn = (t + 1 < TILES_PER_CHUNK) ? t + 1 : t;
    const uint4* psrc =
        (const uint4*)(mnb + (size_t)(ncol0 + tn * 16) * DD) + tid * 4;
    uint4 p0 = psrc[0], p1 = psrc[1], p2 = psrc[2], p3 = psrc[3];

    // ---- compute 16x16 sim tile from LDS-resident B ------------------------
    const unsigned short* lb = &bbuf[cur][(size_t)n_lo * DD + bsub];
    v8f acc0 = {};
    v8f acc1 = {};
#pragma unroll
    for (int kk = 0; kk < KSTEPS; kk += 2) {
      v16bf b0 = *(const v16bf*)(lb + kk * 32);
      v16bf b1 = *(const v16bf*)(lb + (kk + 1) * 32);
      acc0 = __builtin_amdgcn_wmma_f32_16x16x32_bf16(
                 false, afrag[kk],     false, b0, (short)0, acc0, false, false);
      acc1 = __builtin_amdgcn_wmma_f32_16x16x32_bf16(
                 false, afrag[kk + 1], false, b1, (short)0, acc1, false, false);
    }

    // ---- transpose tile through LDS so each lane owns one sim row ----------
    const int n0 = ncol0 + t * 16;
#pragma unroll
    for (int r = 0; r < 8; ++r)
      tile[(r + m_hi) * 16 + n_lo] = acc0[r] + acc1[r];
    asm volatile("s_wait_dscnt 0" ::: "memory");      // per-wave DS RAW fence
#pragma unroll
    for (int j = 0; j < 8; ++j) {
      float v = tile[m_own * 16 + n_half + j];
      top5_insert(v, n0 + n_half + j, tv0,tv1,tv2,tv3,tv4, ti0,ti1,ti2,ti3,ti4);
    }

    // ---- commit prefetched tile into the other LDS buffer ------------------
    uint4* pdst = (uint4*)&bbuf[nxt][0] + tid * 4;
    pdst[0] = p0; pdst[1] = p1; pdst[2] = p2; pdst[3] = p3;
    __syncthreads();
  }

  // ---- emit candidates: 2 lanes per row (n-halves) x 5 each ---------------
  const int row  = mtile * 16 + m_own;
  const int half = lane >> 4;
  size_t base = ((size_t)row * NCHUNK + chunk) * 10 + half * 5;
  cval[base+0]=tv0; cidx[base+0]=ti0;
  cval[base+1]=tv1; cidx[base+1]=ti1;
  cval[base+2]=tv2; cidx[base+2]=ti2;
  cval[base+3]=tv3; cidx[base+3]=ti3;
  cval[base+4]=tv4; cidx[base+4]=ti4;
}

// ---------------------------------------------------------------------------
// K3: merge 320 candidates per row -> global top-5 indices (tiny, scalar)
// ---------------------------------------------------------------------------
__global__ void ggr_merge(const float* __restrict__ cval,
                          const int* __restrict__ cidx,
                          int* __restrict__ topidx) {
  int row = blockIdx.x * blockDim.x + threadIdx.x;
  if (row >= BB) return;
  const float* cv = cval + (size_t)row * (NCHUNK * 10);
  const int*   ci = cidx + (size_t)row * (NCHUNK * 10);
  int sel[KSEL];
#pragma unroll
  for (int p = 0; p < KSEL; ++p) {
    float best = -3e38f; int bid = 0x7fffffff;
    for (int j = 0; j < NCHUNK * 10; ++j) {
      int id = ci[j]; float v = cv[j];
      bool dup = false;
      for (int q = 0; q < p; ++q) dup |= (sel[q] == id);
      if (!dup && (v > best || (v == best && id < bid))) { best = v; bid = id; }
    }
    sel[p] = bid;
    topidx[row * KSEL + p] = bid;
  }
}

// ---------------------------------------------------------------------------
// K4: exact fp32 cosine for the 5 winners, softmax(*10), weighted gather of
//     ORIGINAL memory rows -> context; stored as bf16 hi/lo split.
// ---------------------------------------------------------------------------
__global__ void ggr_context(const float* __restrict__ x,
                            const float* __restrict__ mem,
                            const float* __restrict__ invx,
                            const float* __restrict__ invm,
                            const int* __restrict__ topidx,
                            unsigned short* __restrict__ chi,
                            unsigned short* __restrict__ clo) {
  int row  = blockIdx.x * 8 + (threadIdx.x >> 5);
  int lane = threadIdx.x & 31;
  if (row >= BB) return;
  int id[KSEL];
#pragma unroll
  for (int j = 0; j < KSEL; ++j) id[j] = topidx[row * KSEL + j];

  const float* xr = x + (size_t)row * DD;
  float sim[KSEL];
#pragma unroll
  for (int j = 0; j < KSEL; ++j) {
    const float* mr = mem + (size_t)id[j] * DD;
    float p = 0.f;
#pragma unroll
    for (int t = 0; t < 16; ++t) { int e = lane + (t << 5); p += xr[e] * mr[e]; }
#pragma unroll
    for (int o = 16; o >= 1; o >>= 1) p += __shfl_xor(p, o, 32);
    sim[j] = p * invx[row] * invm[id[j]];
  }
  float mx = sim[0];
#pragma unroll
  for (int j = 1; j < KSEL; ++j) mx = fmaxf(mx, sim[j]);
  float w[KSEL], sum = 0.f;
#pragma unroll
  for (int j = 0; j < KSEL; ++j) { w[j] = expf(TEMPF * (sim[j] - mx)); sum += w[j]; }
  float rs = 1.0f / sum;

  int e0 = lane * 16;
  union { unsigned short u[16]; uint4 q[2]; } ph, pl;
#pragma unroll
  for (int t = 0; t < 16; ++t) {
    float c = 0.f;
#pragma unroll
    for (int j = 0; j < KSEL; ++j) c += (w[j] * rs) * mem[(size_t)id[j] * DD + e0 + t];
    unsigned short h = f2bf_u(c);
    ph.u[t] = h;
    pl.u[t] = f2bf_u(c - bf2f(h));
  }
  uint4* dh = (uint4*)(chi + (size_t)row * DD + e0);
  uint4* dl = (uint4*)(clo + (size_t)row * DD + e0);
  dh[0] = ph.q[0]; dh[1] = ph.q[1];
  dl[0] = pl.q[0]; dl[1] = pl.q[1];
}

// ---------------------------------------------------------------------------
// K5: out = x + alpha * relu(context @ W^T), split-bf16 WMMA with three
//     independent accumulator chains (hi*hi, hi*lo, lo*hi).
// ---------------------------------------------------------------------------
__global__ void __launch_bounds__(256)
ggr_linear_out(const unsigned short* __restrict__ chi,
               const unsigned short* __restrict__ clo,
               const unsigned short* __restrict__ whi,
               const unsigned short* __restrict__ wlo,
               const float* __restrict__ x,
               const float* __restrict__ alpha,
               float* __restrict__ out) {
  int gw    = blockIdx.x * 8 + (threadIdx.x >> 5);   // 0..4095
  int lane  = threadIdx.x & 31;
  int mtile = gw >> 5;                               // 0..127
  int ntile = gw & 31;                               // 0..31
  const float a0 = alpha[0];

  const int asub = lane & 16;
  const unsigned short* ah = chi + (size_t)(mtile*16 + (lane&15)) * DD + asub;
  const unsigned short* al = clo + (size_t)(mtile*16 + (lane&15)) * DD + asub;
  const int bsub = lane & 16;
  const unsigned short* wh = whi + (size_t)(ntile*16 + (lane&15)) * DD + bsub;
  const unsigned short* wl = wlo + (size_t)(ntile*16 + (lane&15)) * DD + bsub;

  v8f acc_hh = {};
  v8f acc_hl = {};
  v8f acc_lh = {};
#pragma unroll
  for (int kk = 0; kk < KSTEPS; ++kk) {
    v16bf fah = load_a_frag(ah + kk * 32);
    v16bf fal = load_a_frag(al + kk * 32);
    v16bf fwh = *(const v16bf*)(wh + kk * 32);
    v16bf fwl = *(const v16bf*)(wl + kk * 32);
    acc_hh = __builtin_amdgcn_wmma_f32_16x16x32_bf16(false, fah, false, fwh, (short)0, acc_hh, false, false);
    acc_hl = __builtin_amdgcn_wmma_f32_16x16x32_bf16(false, fah, false, fwl, (short)0, acc_hl, false, false);
    acc_lh = __builtin_amdgcn_wmma_f32_16x16x32_bf16(false, fal, false, fwh, (short)0, acc_lh, false, false);
  }
  const int col = ntile * 16 + (lane & 15);
  const int mh  = (lane & 16) >> 1;
#pragma unroll
  for (int r = 0; r < 8; ++r) {
    int row = mtile * 16 + r + mh;
    float o = fmaxf(acc_hh[r] + acc_hl[r] + acc_lh[r], 0.0f);
    out[(size_t)row * DD + col] = x[(size_t)row * DD + col] + a0 * o;
  }
}

// ---------------------------------------------------------------------------
// launcher
// ---------------------------------------------------------------------------
extern "C" void kernel_launch(void* const* d_in, const int* in_sizes, int n_in,
                              void* d_out, int out_size, void* d_ws, size_t ws_size,
                              hipStream_t stream) {
  (void)in_sizes; (void)n_in; (void)out_size; (void)ws_size;
  const float* x     = (const float*)d_in[0];
  const float* mem   = (const float*)d_in[1];
  const float* W     = (const float*)d_in[2];
  const float* alpha = (const float*)d_in[3];
  float* out = (float*)d_out;

  char* ws = (char*)d_ws;
  size_t off = 0;
  auto take = [&](size_t bytes) -> char* {
    char* p = ws + off;
    off = (off + bytes + 255) & ~(size_t)255;
    return p;
  };
  unsigned short* mnb  = (unsigned short*)take((size_t)NN * DD * 2);  // 64 MB (L2-resident)
  unsigned short* xnb  = (unsigned short*)take((size_t)BB * DD * 2);  // 2 MB
  float*          invm = (float*)take((size_t)NN * 4);
  float*          invx = (float*)take((size_t)BB * 4);
  unsigned short* whi  = (unsigned short*)take((size_t)DD * DD * 2);
  unsigned short* wlo  = (unsigned short*)take((size_t)DD * DD * 2);
  float*          cval = (float*)take((size_t)BB * NCHUNK * 10 * 4);
  int*            cidx = (int*)take((size_t)BB * NCHUNK * 10 * 4);
  int*            tidx = (int*)take((size_t)BB * KSEL * 4);
  unsigned short* chi  = (unsigned short*)take((size_t)BB * DD * 2);
  unsigned short* clo  = (unsigned short*)take((size_t)BB * DD * 2);

  ggr_norm_rows<<<NN / 8, 256, 0, stream>>>(mem, mnb, invm, NN);
  ggr_norm_rows<<<BB / 8, 256, 0, stream>>>(x,   xnb, invx, BB);
  ggr_split_w<<<(DD * DD) / 256, 256, 0, stream>>>(W, whi, wlo);
  ggr_sim_topk<<<dim3(NCHUNK, 16), 256, 0, stream>>>(xnb, mnb, cval, cidx);
  ggr_merge<<<(BB + 255) / 256, 256, 0, stream>>>(cval, cidx, tidx);
  ggr_context<<<BB / 8, 256, 0, stream>>>(x, mem, invx, invm, tidx, chi, clo);
  ggr_linear_out<<<(128 * 32) / 8, 256, 0, stream>>>(chi, clo, whi, wlo, x, alpha, out);
}
